// MultiHeadAttention_firstlayer_13142599926299
// MI455X (gfx1250) — matrix operand order repaired and checked by reading, not compiled, not run on hardware
//
#include <hip/hip_runtime.h>
#include <hip/hip_bf16.h>

// MultiHeadAttention first layer, d_k == 1.
// softmax is shift invariant and kk[b,i,h] is constant over the softmax axis ->
// attn = softmax_j(posi_att[h,i,:], diag=-inf): independent of q/k/w_k/b_k and
// of batch (broadcast 8x). Pipeline (all GEMMs v_wmma_f32_16x16x32_bf16):
//   K0: bf16-convert v, w_v, w_fc into workspace (streaming, VALU amortized)
//   K1: vvT[b,h,d,l]   = bf16( v @ w_v.T + b_v )      (packed-bf16 A and B)
//   K2: attn[h,b,i,:]  = softmax(posi[h,i,:]), broadcast over b  (512 MB write,
//       the dominant memory cost: ~22 us at 23.3 TB/s)
//   K3: out_pre[b,q,hd]= attn @ vvT  (A cvt on the fly, 16x64 strip per wave)
//   K4: out = LayerNorm( out_pre @ w_fc.T + b_fc )    (fused, LDS row tile)
// Workspace: 52 MB (vvT 16 + out_pre 16 + v_bf 16 + w_v 2 + w_fc 2).

#define HN    16
#define DVV   64
#define DMM   1024
#define BATCH 8
#define SEQ   1024
#define LN_EPS 1e-5f

typedef __attribute__((ext_vector_type(16))) __bf16 v16bf;
typedef __attribute__((ext_vector_type(8)))  float  v8f;

union Frag { v16bf v; unsigned int u[8]; };

__device__ __forceinline__ float wave_max(float x) {
  for (int o = 16; o; o >>= 1) x = fmaxf(x, __shfl_xor(x, o, 32));
  return x;
}
__device__ __forceinline__ float wave_sum(float x) {
  for (int o = 16; o; o >>= 1) x += __shfl_xor(x, o, 32);
  return x;
}

// Pack two f32 -> two bf16 (RNE) letting the compiler pick native cvt ops.
__device__ __forceinline__ unsigned int pk_bf2(float a, float b) {
  union { __bf16 h[2]; unsigned int u; } r;
  r.h[0] = (__bf16)a; r.h[1] = (__bf16)b;
  return r.u;
}
__device__ __forceinline__ unsigned short bf1(float a) {
  union { __bf16 h; unsigned short s; } r;
  r.h = (__bf16)a;
  return r.s;
}

// --- WMMA fragment loads from pre-packed bf16 (layouts per CDNA5 ISA 7.12.2) --
// A (16x32): lane m=lane&15; K groups {hi*8..hi*8+7} and {16+hi*8..+7} are
// contiguous -> 2 x b128 loads.
__device__ __forceinline__ void ldA_bf(Frag& f, const unsigned short* p, int kb, int hi) {
  *(uint4*)&f.u[0] = *(const uint4*)(p + kb + hi * 8);
  *(uint4*)&f.u[4] = *(const uint4*)(p + kb + 16 + hi * 8);
}
// B (32x16): lane n=lane&15; K groups {hi*8..hi*8+7} and {16+hi*8..+7} (same as A) -> 2 x b128 loads.
__device__ __forceinline__ void ldB_bf(Frag& f, const unsigned short* p, int kb, int hi) {
  *(uint4*)&f.u[0] = *(const uint4*)(p + kb + hi * 8);
  *(uint4*)&f.u[4] = *(const uint4*)(p + kb + 16 + hi * 8);
}
// A fragment from fp32 with on-the-fly bf16 conversion (4 x b128 + pk cvts).
__device__ __forceinline__ void cvtA_f32(Frag& f, const float* p, int kb, int hi) {
  const float4 q0 = *(const float4*)(p + kb + hi * 8);
  const float4 q1 = *(const float4*)(p + kb + hi * 8 + 4);
  const float4 q2 = *(const float4*)(p + kb + 16 + hi * 8);
  const float4 q3 = *(const float4*)(p + kb + 16 + hi * 8 + 4);
  f.u[0] = pk_bf2(q0.x, q0.y); f.u[1] = pk_bf2(q0.z, q0.w);
  f.u[2] = pk_bf2(q1.x, q1.y); f.u[3] = pk_bf2(q1.z, q1.w);
  f.u[4] = pk_bf2(q2.x, q2.y); f.u[5] = pk_bf2(q2.z, q2.w);
  f.u[6] = pk_bf2(q3.x, q3.y); f.u[7] = pk_bf2(q3.z, q3.w);
}

// ------------------- K0: streaming f32 -> packed bf16 --------------------------
__global__ __launch_bounds__(256)
void cvt_bf16_kernel(const float* __restrict__ src, unsigned short* __restrict__ dst,
                     int n4) {
  const int i = blockIdx.x * 256 + threadIdx.x;
  if (i < n4) {
    const float4 f = ((const float4*)src)[i];
    uint2 o;
    o.x = pk_bf2(f.x, f.y);
    o.y = pk_bf2(f.z, f.w);
    ((uint2*)dst)[i] = o;
  }
}

// ------------- K1: vvT[b,h,d,l] = bf16( v @ w_v.T + b_v ) ----------------------
// Wave computes a 16(m) x 64(n) strip: one A fragment feeds 4 WMMAs.
__global__ __launch_bounds__(256)
void vv_proj_kernel(const unsigned short* __restrict__ v_bf,
                    const unsigned short* __restrict__ w_v_bf,
                    const float* __restrict__ b_v,
                    unsigned short* __restrict__ vvT) {
  const int lane = threadIdx.x & 31, wave = threadIdx.x >> 5;
  const int tile = blockIdx.x * 8 + wave;
  const int ng = tile & 15;            // 16 n-groups of 64 over H*DV; ng == head
  const int mtile = tile >> 4;         // 512 m-tiles over B*SEQ
  const int hi = lane >> 4, lo = lane & 15;

  const unsigned short* Arow = v_bf + (size_t)(mtile * 16 + lo) * DMM;
  const unsigned short* Bbase = w_v_bf + (size_t)(ng * 64 + lo) * DMM;

  v8f acc[4] = {};
  for (int kb = 0; kb < DMM; kb += 32) {
    Frag a;
    ldA_bf(a, Arow, kb, hi);
    __builtin_prefetch(Arow + kb + 64, 0, 0);
#pragma unroll
    for (int t = 0; t < 4; ++t) {
      Frag b;
      ldB_bf(b, Bbase + (size_t)t * 16 * DMM, kb, hi);
      acc[t] = __builtin_amdgcn_wmma_f32_16x16x32_bf16(false, a.v, false, b.v,
                                                       (short)0, acc[t], false, false);
    }
  }
  const int r0 = mtile * 16 + hi * 8;         // first (b,l) row of this thread
  const int bi = r0 >> 10, l0 = r0 & 1023;    // 16-row tiles never straddle batches
#pragma unroll
  for (int t = 0; t < 4; ++t) {
    const int d = t * 16 + lo;
    const float bias = b_v[ng * 64 + d];
    uint4 o;                                  // 8 consecutive l's -> one b128 store
    o.x = pk_bf2(acc[t][0] + bias, acc[t][1] + bias);
    o.y = pk_bf2(acc[t][2] + bias, acc[t][3] + bias);
    o.z = pk_bf2(acc[t][4] + bias, acc[t][5] + bias);
    o.w = pk_bf2(acc[t][6] + bias, acc[t][7] + bias);
    *(uint4*)(vvT + ((((size_t)bi * HN + ng) * DVV + d) * SEQ + l0)) = o;
  }
}

// ------ K2: attn[h,b,i,:] = softmax(posi[h,i,:], diag=-inf), dup over b --------
__global__ __launch_bounds__(256)
void softmax_bcast_kernel(const float* __restrict__ posi, float* __restrict__ attn) {
  const int lane = threadIdx.x & 31, wave = threadIdx.x >> 5;
  const int row = blockIdx.x * 8 + wave;      // h*SEQ + i
  const int h = row >> 10, i = row & 1023;
  const float* src = posi + (size_t)row * SEQ;

  float vb[32];
  float m = -INFINITY;
#pragma unroll
  for (int t = 0; t < 32; ++t) {
    const int j = t * 32 + lane;
    float x = src[j];
    if (j == i) x = -INFINITY;                // masked diagonal
    vb[t] = x;
    m = fmaxf(m, x);
  }
  m = wave_max(m);
  float s = 0.f;
#pragma unroll
  for (int t = 0; t < 32; ++t) { vb[t] = __expf(vb[t] - m); s += vb[t]; }
  s = wave_sum(s);
  const float inv = 1.0f / s;
#pragma unroll
  for (int t = 0; t < 32; ++t) vb[t] *= inv;
  for (int b = 0; b < BATCH; ++b) {
    float* dst = attn + (((size_t)h * BATCH + b) * SEQ + i) * SEQ;
#pragma unroll
    for (int t = 0; t < 32; ++t) dst[t * 32 + lane] = vb[t];
  }
}

// ------ K3: out_pre[b,q,h*64+d] = bf16( sum_k attn[h,b,q,k] * vvT[b,h,d,k] ) ---
// Wave computes 16(q) x 64(d): one A fragment (softmax row block) feeds 4 WMMAs.
__global__ __launch_bounds__(256)
void av_gemm_kernel(const float* __restrict__ attn,
                    const unsigned short* __restrict__ vvT,
                    unsigned short* __restrict__ out_pre) {
  const int lane = threadIdx.x & 31, wave = threadIdx.x >> 5;
  const int tile = blockIdx.x * 8 + wave;     // b(8) x h(16) x qtile(64)
  const int qtile = tile & 63;
  const int h = (tile >> 6) & 15;
  const int b = tile >> 10;
  const int hi = lane >> 4, lo = lane & 15;

  const float* Arow = attn + (((size_t)h * BATCH + b) * SEQ + qtile * 16 + lo) * SEQ;
  const unsigned short* Bbase = vvT + (((size_t)b * HN + h) * DVV + lo) * SEQ;

  v8f acc[4] = {};
  for (int kb = 0; kb < SEQ; kb += 32) {
    Frag a;
    cvtA_f32(a, Arow, kb, hi);
    __builtin_prefetch(Arow + kb + 64, 0, 0);
#pragma unroll
    for (int t = 0; t < 4; ++t) {
      Frag bf;
      ldB_bf(bf, Bbase + (size_t)t * 16 * SEQ, kb, hi);
      acc[t] = __builtin_amdgcn_wmma_f32_16x16x32_bf16(false, a.v, false, bf.v,
                                                       (short)0, acc[t], false, false);
    }
  }
#pragma unroll
  for (int t = 0; t < 4; ++t) {
    const int n = h * DVV + t * 16 + lo;
#pragma unroll
    for (int j = 0; j < 8; ++j) {
      const int q = qtile * 16 + hi * 8 + j;
      out_pre[((size_t)b * SEQ + q) * (HN * DVV) + n] = bf1(acc[t][j]);
    }
  }
}

// ---------- K4: out = LayerNorm( out_pre @ w_fc.T + b_fc ) (fused) -------------
__global__ __launch_bounds__(256)
void fc_ln_kernel(const unsigned short* __restrict__ x_bf,
                  const unsigned short* __restrict__ w_fc_bf,
                  const float* __restrict__ b_fc, const float* __restrict__ ln_g,
                  const float* __restrict__ ln_b, float* __restrict__ out) {
  __shared__ float tile[16 * DMM];            // 64 KB of the 320 KB WGP LDS
  const int lane = threadIdx.x & 31, wave = threadIdx.x >> 5;
  const int hi = lane >> 4, lo = lane & 15;
  const int r0 = blockIdx.x * 16;

  const unsigned short* Arow = x_bf + (size_t)(r0 + lo) * DMM;
  for (int nt = wave; nt < DMM / 16; nt += 8) {
    const int nG = nt * 16 + lo;
    const unsigned short* Brow = w_fc_bf + (size_t)nG * DMM;
    v8f acc = {};
    for (int kb = 0; kb < DMM; kb += 32) {
      Frag a, bf;
      ldA_bf(a, Arow, kb, hi);
      ldB_bf(bf, Brow, kb, hi);
      acc = __builtin_amdgcn_wmma_f32_16x16x32_bf16(false, a.v, false, bf.v,
                                                    (short)0, acc, false, false);
    }
    const float bias = b_fc[nG];
#pragma unroll
    for (int j = 0; j < 8; ++j)
      tile[(hi * 8 + j) * DMM + nG] = acc[j] + bias;
  }
  __syncthreads();

  // LayerNorm over DM=1024 per row; 8 waves x 2 rows.
  for (int lr = wave * 2; lr < wave * 2 + 2; ++lr) {
    float s = 0.f, ss = 0.f;
#pragma unroll 4
    for (int t = 0; t < DMM / 32; ++t) {
      const float xv = tile[lr * DMM + t * 32 + lane];
      s += xv; ss += xv * xv;
    }
    s = wave_sum(s); ss = wave_sum(ss);
    const float mu = s * (1.0f / DMM);
    const float var = ss * (1.0f / DMM) - mu * mu;  // ddof=0, matches jnp.var
    const float rs = rsqrtf(var + LN_EPS);
    float* dst = out + (size_t)(r0 + lr) * DMM;
#pragma unroll 4
    for (int t = 0; t < DMM / 32; ++t) {
      const int j = t * 32 + lane;
      dst[j] = (tile[lr * DMM + j] - mu) * rs * ln_g[j] + ln_b[j];
    }
  }
}

extern "C" void kernel_launch(void* const* d_in, const int* in_sizes, int n_in,
                              void* d_out, int out_size, void* d_ws, size_t ws_size,
                              hipStream_t stream) {
  // setup_inputs order: q,k,v,posi_att,w_k,b_k,w_v,b_v,w_fc,b_fc,ln_g,ln_b
  // q,k,w_k,b_k are mathematically dead (softmax shift invariance) -> unused.
  const float* v    = (const float*)d_in[2];
  const float* posi = (const float*)d_in[3];
  const float* w_v  = (const float*)d_in[6];
  const float* b_v  = (const float*)d_in[7];
  const float* w_fc = (const float*)d_in[8];
  const float* b_fc = (const float*)d_in[9];
  const float* ln_g = (const float*)d_in[10];
  const float* ln_b = (const float*)d_in[11];

  float* out  = (float*)d_out;                                 // [B,L,DM]
  float* attn = (float*)d_out + (size_t)BATCH * SEQ * DMM;     // [H,B,L,L]

  // Workspace layout (ushort units): 52 MB total.
  unsigned short* vvT    = (unsigned short*)d_ws;              // [B,H,DV,L]  8M
  unsigned short* opre   = vvT  + (size_t)8 * 1024 * 1024;     // [B,L,H*DV]  8M
  unsigned short* v_bf   = opre + (size_t)8 * 1024 * 1024;     // [B*L,DM]    8M
  unsigned short* wv_bf  = v_bf + (size_t)8 * 1024 * 1024;     // [H*DV,DM]   1M
  unsigned short* wfc_bf = wv_bf + (size_t)1024 * 1024;        // [DM,H*DV]   1M

  cvt_bf16_kernel<<<(2 * 1024 * 1024) / 256, 256, 0, stream>>>(v, v_bf, 2 * 1024 * 1024);
  cvt_bf16_kernel<<<(256 * 1024) / 256, 256, 0, stream>>>(w_v, wv_bf, 256 * 1024);
  cvt_bf16_kernel<<<(256 * 1024) / 256, 256, 0, stream>>>(w_fc, wfc_bf, 256 * 1024);

  vv_proj_kernel      <<<(512 * 16) / 8, 256, 0, stream>>>(v_bf, wv_bf, b_v, vvT);
  softmax_bcast_kernel<<<(HN * SEQ) / 8, 256, 0, stream>>>(posi, attn);
  av_gemm_kernel      <<<(BATCH * HN * 64) / 8, 256, 0, stream>>>(attn, vvT, opre);
  fc_ln_kernel        <<<(BATCH * SEQ) / 16, 256, 0, stream>>>(opre, wfc_bf, b_fc,
                                                               ln_g, ln_b, out);
}
